// AttentionHead_27350351741618
// MI455X (gfx1250) — matrix operand order repaired and checked by reading, not compiled, not run on hardware
//
#include <hip/hip_runtime.h>
#include <hip/hip_bf16.h>

typedef __attribute__((ext_vector_type(16))) _Float16 v16h;
typedef __attribute__((ext_vector_type(8)))  _Float16 v8h;
typedef __attribute__((ext_vector_type(8)))  float    v8f;

#define BATCH 4
#define SEQ   4096
#define DIM   128
#define WAVES 4          // 128 threads/block
#define NEG_BIG (-1.0e30f)

__device__ __forceinline__ v8f wmma_f16(const v16h& a, const v16h& b, const v8f& c) {
    return __builtin_amdgcn_wmma_f32_16x16x32_f16(false, a, false, b, (short)0, c, false, false);
}

// ---------------------------------------------------------------------------
// Pass 0: pack W_Q/W_K/W_V (f32, row-major DxD) into f16 B-fragment order:
//   Wp[((mat*8 + nt)*4 + ks)*32 + lane][i] = W[ks*32 + (lane>>4)*16 + i][nt*16 + (lane&15)]
// One thread emits one lane's 16 contiguous halfs. Strided gather paid ONCE.
// ---------------------------------------------------------------------------
__global__ __launch_bounds__(128)
void pack_w_kernel(const float* __restrict__ Wq,
                   const float* __restrict__ Wk,
                   const float* __restrict__ Wv,
                   _Float16* __restrict__ Wp) {
    int gid = blockIdx.x * blockDim.x + threadIdx.x;   // 0 .. 3*8*4*32-1
    if (gid >= 3 * 8 * 4 * 32) return;
    const int lane = gid & 31;
    int t = gid >> 5;
    const int ks = t & 3;  t >>= 2;
    const int nt = t & 7;  t >>= 3;
    const int mat = t;                                  // 0..2
    const float* W = (mat == 0) ? Wq : (mat == 1) ? Wk : Wv;
    const int n  = lane & 15;
    const int hf = lane >> 4;
    const float* src = W + (size_t)(ks * 32 + hf * 16) * DIM + nt * 16 + n;
    _Float16* dst = Wp + (size_t)gid * 16;
#pragma unroll
    for (int i = 0; i < 16; ++i) dst[i] = (_Float16)src[(size_t)i * DIM];
}

// ---------------------------------------------------------------------------
// Pass 1: grid = (strips/WAVES, 3); blockIdx.y = matrix (SGPR-uniform so the
// output-path branch is a scalar branch, no exec juggling).
// One wave computes a 16x128 strip: A-fragments loaded once, 8 col tiles x
// 4 k-steps = 32 WMMAs. B-fragments are contiguous 32B loads from Wp.
// ---------------------------------------------------------------------------
__global__ __launch_bounds__(WAVES * 32)
void proj_kernel(const float* __restrict__ emb,
                 const _Float16* __restrict__ Wp,
                 _Float16* __restrict__ Qh,
                 _Float16* __restrict__ Kh,
                 _Float16* __restrict__ Vt) {
    const int lane = threadIdx.x & 31;
    const int wave = threadIdx.x >> 5;
    const int m    = lane & 15;
    const int hf   = lane >> 4;

    const int mat = blockIdx.y;                       // 0=Q, 1=K, 2=V (scalar)
    int strip = blockIdx.x * WAVES + wave;
    if (strip >= BATCH * (SEQ / 16)) return;
    const int b  = strip / (SEQ / 16);
    const int mt = strip % (SEQ / 16);

    // Resident A-fragments: emb rows mt*16..+15, f32 -> f16, 4 k-steps of 16x32
    const float* A = emb + (size_t)b * SEQ * DIM + (size_t)mt * 16 * DIM;
    v16h af[4];
#pragma unroll
    for (int ks = 0; ks < 4; ++ks) {
        const float* ap = A + m * DIM + ks * 32 + hf * 16;
#pragma unroll
        for (int i = 0; i < 8; ++i) { af[ks][i] = (_Float16)ap[i]; af[ks][8 + i] = (_Float16)ap[8 + i]; }
    }

    const float scale = (mat == 0) ? 0.08838834764831845f : 1.0f;  // 1/sqrt(128)
    _Float16* qk = ((mat == 0) ? Qh : Kh) + (size_t)b * SEQ * DIM;
    _Float16* vt = Vt + (size_t)b * DIM * SEQ;

#pragma unroll
    for (int nt = 0; nt < 8; ++nt) {
        v8f acc = {};
#pragma unroll
        for (int ks = 0; ks < 4; ++ks) {
            v16h bf = *(const v16h*)(Wp + ((((size_t)mat * 8 + nt) * 4 + ks) * 32 + lane) * 16);
            acc = wmma_f16(af[ks], bf, acc);
        }
        if (mat == 2) {
            // V transposed: Vt[b][d][s] so PV B-fragments are contiguous
#pragma unroll
            for (int v = 0; v < 8; ++v) {
                const int row = mt * 16 + v + 8 * hf;   // key index
                const int col = nt * 16 + m;            // d index
                vt[(size_t)col * SEQ + row] = (_Float16)acc[v];
            }
        } else {
#pragma unroll
            for (int v = 0; v < 8; ++v) {
                const int row = mt * 16 + v + 8 * hf;
                qk[(size_t)row * DIM + nt * 16 + m] = (_Float16)(acc[v] * scale);
            }
        }
    }
}

// ---------------------------------------------------------------------------
// Pass 2: flash attention. grid = (S/16/WAVES, BATCH); blockIdx.y = batch.
// One wave owns a 16-query tile, streams 32-key blocks up to the causal
// boundary: 16 WMMAs/block, prefetch of the next K/V tiles overlapped.
// ---------------------------------------------------------------------------
__global__ __launch_bounds__(WAVES * 32)
void attn_kernel(const _Float16* __restrict__ Qh,
                 const _Float16* __restrict__ Kh,
                 const _Float16* __restrict__ Vt,
                 float* __restrict__ out) {
    __shared__ _Float16 plds[WAVES][16 * 32];   // per-wave P tile (16q x 32k)

    const int lane = threadIdx.x & 31;
    const int wave = threadIdx.x >> 5;
    const int m    = lane & 15;
    const int hf   = lane >> 4;

    const int b = blockIdx.y;                        // scalar batch index
    const int q0 = (blockIdx.x * WAVES + wave) * 16;
    if (q0 >= SEQ) return;

    const _Float16* Qb = Qh + (size_t)b * SEQ * DIM;
    const _Float16* Kb = Kh + (size_t)b * SEQ * DIM;
    const _Float16* Vb = Vt + (size_t)b * DIM * SEQ;
    _Float16* myp = &plds[wave][0];

    // Resident Q fragments: 4 k-steps of 16x32 (already pre-scaled by 1/sqrt(D))
    v16h qf[4];
#pragma unroll
    for (int ks = 0; ks < 4; ++ks) {
        const _Float16* p = Qb + (size_t)(q0 + m) * DIM + ks * 32 + hf * 16;
        v8h lo = *(const v8h*)p;
        v8h hi = *(const v8h*)(p + 8);
#pragma unroll
        for (int i = 0; i < 8; ++i) { qf[ks][i] = lo[i]; qf[ks][8 + i] = hi[i]; }
    }

    v8f  o[8];
    float mrow[8], lrow[8];
#pragma unroll
    for (int t = 0; t < 8; ++t) { o[t] = (v8f){}; mrow[t] = NEG_BIG; lrow[t] = 0.0f; }

    const int kb_last = ((q0 + 15) >> 5) << 5;
    for (int kb = 0; kb <= kb_last; kb += 32) {
        // ---- prefetch next key block (speculative; dropped if OOB) ----
        {
            const _Float16* knext = Kb + (size_t)(kb + 32 + lane) * DIM;
            __builtin_prefetch(knext, 0, 3);
            __builtin_prefetch(knext + 64, 0, 3);        // second half of 256B row
#pragma unroll
            for (int j = 0; j < 4; ++j)
                __builtin_prefetch(Vb + (size_t)(j * 32 + lane) * SEQ + kb + 32, 0, 3);
        }

        // ---- S = Q @ K^T for 16q x 32k ----
        v8f s0 = {}, s1 = {};
#pragma unroll
        for (int ks = 0; ks < 4; ++ks) {
            const int d0 = ks * 32 + hf * 16;
            v16h b0 = *(const v16h*)(Kb + (size_t)(kb + m) * DIM + d0);
            v16h b1 = *(const v16h*)(Kb + (size_t)(kb + 16 + m) * DIM + d0);
            s0 = wmma_f16(qf[ks], b0, s0);
            s1 = wmma_f16(qf[ks], b1, s1);
        }

        // ---- causal mask (diagonal block only) ----
        if (kb == kb_last) {
#pragma unroll
            for (int v = 0; v < 8; ++v) {
                const int row = q0 + v + 8 * hf;
                if (kb + m      > row) s0[v] = NEG_BIG;
                if (kb + 16 + m > row) s1[v] = NEG_BIG;
            }
        }

        // ---- online softmax: rows live across 16-lane groups ----
        float alpha[8];
#pragma unroll
        for (int v = 0; v < 8; ++v) {
            float mx = fmaxf(s0[v], s1[v]);
            mx = fmaxf(mx, __shfl_xor(mx, 1, 32));
            mx = fmaxf(mx, __shfl_xor(mx, 2, 32));
            mx = fmaxf(mx, __shfl_xor(mx, 4, 32));
            mx = fmaxf(mx, __shfl_xor(mx, 8, 32));
            const float mnew = fmaxf(mrow[v], mx);
            const float p0 = __expf(s0[v] - mnew);
            const float p1 = __expf(s1[v] - mnew);
            float rs = p0 + p1;
            rs += __shfl_xor(rs, 1, 32);
            rs += __shfl_xor(rs, 2, 32);
            rs += __shfl_xor(rs, 4, 32);
            rs += __shfl_xor(rs, 8, 32);
            const float a = __expf(mrow[v] - mnew);
            lrow[v] = lrow[v] * a + rs;
            mrow[v] = mnew;
            alpha[v] = a;
            // stage P (f16) in LDS, C-layout -> [row][key] matrix order
            const int row = v + 8 * hf;
            myp[row * 32 + m]      = (_Float16)p0;
            myp[row * 32 + 16 + m] = (_Float16)p1;
        }
        asm volatile("s_wait_dscnt 0x0" ::: "memory");

        // ---- read P back as a 16x32 A-fragment ----
        v16h pf;
        {
            const _Float16* pp = myp + m * 32 + hf * 16;
            v8h lo = *(const v8h*)pp;
            v8h hi = *(const v8h*)(pp + 8);
#pragma unroll
            for (int i = 0; i < 8; ++i) { pf[i] = lo[i]; pf[8 + i] = hi[i]; }
        }

        // ---- O = diag(alpha)*O + P @ V  (V^T rows are contiguous) ----
#pragma unroll
        for (int dt = 0; dt < 8; ++dt) {
            v16h vf = *(const v16h*)(Vb + (size_t)(dt * 16 + m) * SEQ + kb + hf * 16);
            v8f c;
#pragma unroll
            for (int v = 0; v < 8; ++v) c[v] = o[dt][v] * alpha[v];
            o[dt] = wmma_f16(pf, vf, c);
        }
        asm volatile("" ::: "memory");   // keep next iter's LDS stores after the loads
    }

    // ---- epilogue: O / l ----
    float inv[8];
#pragma unroll
    for (int v = 0; v < 8; ++v) inv[v] = 1.0f / lrow[v];
    float* ob = out + (size_t)b * SEQ * DIM;
#pragma unroll
    for (int dt = 0; dt < 8; ++dt)
#pragma unroll
        for (int v = 0; v < 8; ++v) {
            const int row = q0 + v + 8 * hf;
            ob[(size_t)row * DIM + dt * 16 + m] = o[dt][v] * inv[v];
        }
}

extern "C" void kernel_launch(void* const* d_in, const int* in_sizes, int n_in,
                              void* d_out, int out_size, void* d_ws, size_t ws_size,
                              hipStream_t stream) {
    const float* emb = (const float*)d_in[0];
    const float* Wq  = (const float*)d_in[1];
    const float* Wk  = (const float*)d_in[2];
    const float* Wv  = (const float*)d_in[3];
    float* out = (float*)d_out;

    const size_t BSD = (size_t)BATCH * SEQ * DIM;
    _Float16* Qh = (_Float16*)d_ws;
    _Float16* Kh = Qh + BSD;
    _Float16* Vt = Kh + BSD;
    _Float16* Wp = Vt + BSD;            // 3*128*128 halfs = 96 KB

    // Pass 0: pack weights into B-fragment order (3*8*4*32 = 3072 threads)
    pack_w_kernel<<<24, 128, 0, stream>>>(Wq, Wk, Wv, Wp);

    // Pass 1: grid.y = matrix (scalar-uniform), one wave per 16-row strip
    {
        dim3 grid((BATCH * (SEQ / 16) + WAVES - 1) / WAVES, 3);
        proj_kernel<<<grid, WAVES * 32, 0, stream>>>(emb, Wp, Qh, Kh, Vt);
    }

    // Pass 2: grid.y = batch, one wave per 16-query tile
    {
        dim3 grid((SEQ / 16 + WAVES - 1) / WAVES, BATCH);
        attn_kernel<<<grid, WAVES * 32, 0, stream>>>(Qh, Kh, Vt, out);
    }
}
